// MultiHeadAttention_31516470018132
// MI455X (gfx1250) — matrix operand order repaired and checked by reading, not compiled, not run on hardware
//
#include <hip/hip_runtime.h>
#include <math.h>
#include <stdint.h>

// ---------------------------------------------------------------------------
// Types for gfx1250 WMMA (wave32): bf16 A/B fragments = 16 x __bf16, C/D = 8 x f32
// ---------------------------------------------------------------------------
typedef __bf16 bf16;
typedef __attribute__((ext_vector_type(16))) __bf16 bf16x16;
typedef __attribute__((ext_vector_type(8)))  __bf16 bf16x8;
typedef __attribute__((ext_vector_type(4)))  __bf16 bf16x4;
typedef __attribute__((ext_vector_type(8)))  float  f32x8;
typedef __attribute__((ext_vector_type(4)))  float  f32x4;
// TDM descriptor groups (this toolchain: 6-arg builtin signature)
typedef __attribute__((ext_vector_type(4))) unsigned int u32x4;
typedef __attribute__((ext_vector_type(8))) int          i32x8;
typedef __attribute__((ext_vector_type(4))) int          i32x4;

#define WMMA_BF16(a, b, c) \
  __builtin_amdgcn_wmma_f32_16x16x32_bf16(false, (a), false, (b), (short)0, (c), false, false)

static __device__ __forceinline__ bf16x16 cat8(bf16x8 lo, bf16x8 hi) {
  return __builtin_shufflevector(lo, hi, 0, 1, 2, 3, 4, 5, 6, 7, 8, 9, 10, 11, 12, 13, 14, 15);
}

// ---------------------------------------------------------------------------
// GEMM: Out[M,N] = A[M,K] @ W[K,N] + bias  (A fp32 or bf16; Out bf16 or fp32)
// Block tile 128x128, 8 waves (4Mx2N), each wave 32x64 (2x4 WMMA tiles), K-step 32.
// ---------------------------------------------------------------------------
template <bool A_BF16, bool OUT_BF16>
__global__ __launch_bounds__(256) void gemm_wmma(
    const void* __restrict__ Aptr, const float* __restrict__ W,
    const float* __restrict__ bias, void* __restrict__ Out,
    int M, int N, int K) {
  constexpr int LDA = 40;                     // LDS row stride in bf16 (32 + 8 pad)
  __shared__ bf16 As[128 * LDA];              // 10240 B
  __shared__ bf16 Bs[128 * LDA];              // 10240 B

  const int t = threadIdx.x;
  const int m0 = blockIdx.y * 128;
  const int n0 = blockIdx.x * 128;
  const int w = t >> 5, l = t & 31;
  const int wm = (w >> 1) * 32, wn = (w & 1) * 64;
  const int lm = l & 15;
  const int half = l >> 4;                    // 0: lanes 0-15, 1: lanes 16-31
  const int k0a = half * 8;                   // A-frag K base (per ISA layout)
  const int k0b = half * 8;                   // B-frag K base

  f32x8 acc[2][4] = {};

  for (int kk = 0; kk < K; kk += 32) {
    // ---- stage A tile (128 x 32) into LDS as bf16 ----
    if (A_BF16) {
      const bf16* A = (const bf16*)Aptr;
#pragma unroll
      for (int it = 0; it < 2; ++it) {
        int q = t + it * 256;                 // 512 x vec8
        int row = q >> 2, c8 = (q & 3) * 8;
        bf16x8 v = *(const bf16x8*)(A + (size_t)(m0 + row) * K + kk + c8);
        *(bf16x8*)(&As[row * LDA + c8]) = v;
      }
    } else {
      const float* A = (const float*)Aptr;
#pragma unroll
      for (int it = 0; it < 4; ++it) {
        int q = t + it * 256;                 // 1024 x vec4
        int row = q >> 3, c4 = (q & 7) * 4;
        f32x4 v = *(const f32x4*)(A + (size_t)(m0 + row) * K + kk + c4);
        bf16x4 b4 = {(bf16)v[0], (bf16)v[1], (bf16)v[2], (bf16)v[3]};
        *(bf16x4*)(&As[row * LDA + c4]) = b4;
      }
    }
    // ---- stage W tile (32k x 128n), transposed -> Bs[n][k] ----
#pragma unroll
    for (int it = 0; it < 4; ++it) {
      int q = t + it * 256;                   // 1024 x vec4
      int krow = q >> 5, c4 = (q & 31) * 4;
      f32x4 v = *(const f32x4*)(W + (size_t)(kk + krow) * N + n0 + c4);
#pragma unroll
      for (int j = 0; j < 4; ++j) Bs[(c4 + j) * LDA + krow] = (bf16)v[j];
    }
    __syncthreads();

    // ---- fragments + 8 WMMA per wave per K-step ----
    bf16x16 af[2], bfg[4];
#pragma unroll
    for (int i = 0; i < 2; ++i) {
      const bf16* p = &As[(wm + i * 16 + lm) * LDA + k0a];
      af[i] = cat8(*(const bf16x8*)p, *(const bf16x8*)(p + 16));
    }
#pragma unroll
    for (int j = 0; j < 4; ++j) {
      const bf16* p = &Bs[(wn + j * 16 + lm) * LDA + k0b];
      bfg[j] = cat8(*(const bf16x8*)p, *(const bf16x8*)(p + 16));
    }
#pragma unroll
    for (int i = 0; i < 2; ++i)
#pragma unroll
      for (int j = 0; j < 4; ++j) acc[i][j] = WMMA_BF16(af[i], bfg[j], acc[i][j]);
    __syncthreads();
  }

  // ---- epilogue: bias + store ----
#pragma unroll
  for (int i = 0; i < 2; ++i) {
#pragma unroll
    for (int j = 0; j < 4; ++j) {
      int col = n0 + wn + j * 16 + lm;
      float bv = bias[col];
#pragma unroll
      for (int r = 0; r < 8; ++r) {
        int row = m0 + wm + i * 16 + r + half * 8;
        float v = acc[i][j][r] + bv;
        if (OUT_BF16)
          ((bf16*)Out)[(size_t)row * N + col] = (bf16)v;
        else
          ((float*)Out)[(size_t)row * N + col] = v;
      }
    }
  }
}

// ---------------------------------------------------------------------------
// Causal flash attention. qkv bf16 [B*S][2304] = [q(768) | k(768) | v(768)],
// head h occupies a contiguous 256-slice. Output bf16 [B*S][768].
// Block: 8 waves x 16 query rows = 128 rows per (b,h). KV tile = 32 keys.
// K tile staged via the Tensor Data Mover (hardware-padded LDS rows); V staged
// manually because it needs a transpose.
// ---------------------------------------------------------------------------
#define S_LEN 2048
#define DH 256
#define QKV_LD 2304

__global__ __launch_bounds__(256) void flash_attn(
    const bf16* __restrict__ qkv, bf16* __restrict__ attn_out) {
  constexpr int KLD = 264;                    // Ks row stride (256 d + 8 pad)
  constexpr int VLD = 40;                     // Vt row stride (32 keys + 8 pad)
  constexpr int PLD = 40;                     // P scratch row stride
  __shared__ bf16 Ks[32 * KLD];               // 16896 B  Ks[key][d]
  __shared__ bf16 Vt[DH * VLD];               // 20480 B  Vt[d][key]
  __shared__ bf16 Ps[8 * 16 * PLD];           // 10240 B  per-wave P scratch

  const int t = threadIdx.x, w = t >> 5, l = t & 31;
  const int bh = blockIdx.y;                  // 0..11
  const int b = bh / 3, h = bh % 3;
  const int r0_blk = blockIdx.x * 128;
  const int r0w = r0_blk + w * 16;
  const int lm = l & 15, half = l >> 4;

  const bf16* qbase = qkv + (size_t)(b * S_LEN) * QKV_LD + h * DH;
  const bf16* kbase = qbase + 768;
  const bf16* vbase = qbase + 1536;

  // ---- preload Q fragments (16 rows x 256 d = 8 x 16x32 frags) ----
  bf16x16 qf[8];
  {
    const bf16* qr = qbase + (size_t)(r0w + lm) * QKV_LD;
#pragma unroll
    for (int ks = 0; ks < 8; ++ks) {
      const bf16* p = qr + ks * 32 + half * 8;
      qf[ks] = cat8(*(const bf16x8*)p, *(const bf16x8*)(p + 16));
    }
  }

  f32x8 accO[16] = {};                        // 16 rows x 256 d accumulator
  float m_run[8], l_run[8];
#pragma unroll
  for (int r = 0; r < 8; ++r) { m_run[r] = -3.0e38f; l_run[r] = 0.0f; }

  const int ntiles = (r0_blk + 128) / 32;     // causal bound for this block
  const float scale = 0.0625f;                // 1/sqrt(256)
  const unsigned int ks_lds = (unsigned int)(uintptr_t)(void*)Ks;  // LDS byte offset

  for (int jt = 0; jt < ntiles; ++jt) {
    const int j0 = jt * 32;

    // ---- TDM: K tile (32 keys x 256 d, row stride 2304) -> Ks, padded rows ----
    if (w == 0) {
      uint64_t ga = (uint64_t)(uintptr_t)(kbase + (size_t)j0 * QKV_LD);
      u32x4 g0;
      g0[0] = 1u;                             // count=1, user descriptor
      g0[1] = ks_lds;                         // lds_addr
      g0[2] = (unsigned int)ga;               // global_addr[31:0]
      g0[3] = (unsigned int)(ga >> 32) | (2u << 30);  // addr[56:32] | type=2
      i32x8 g1;
      g1[0] = (int)((1u << 16)                // data_size = 2 bytes
                  | (1u << 20)                // pad_enable
                  | (6u << 22)                // pad_interval: 128 DWORDs (512B row)
                  | (3u << 25));              // pad_amount: 4 DWORDs (16B pad)
      g1[1] = (int)(256u << 16);              // tensor_dim0 = 256 (low 16)
      g1[2] = (int)(32u << 16);               // tensor_dim0 hi=0 | tensor_dim1 = 32
      g1[3] = (int)(256u << 16);              // tensor_dim1 hi=0 | tile_dim0 = 256
      g1[4] = (int)32u;                       // tile_dim1 = 32, tile_dim2 = 0
      g1[5] = (int)(unsigned int)QKV_LD;      // tensor_dim0_stride = 2304 (lo 32)
      g1[6] = 0;                              // stride hi | tensor_dim1_stride lo
      g1[7] = 0;
      i32x4 gz  = {0, 0, 0, 0};
      i32x8 gz8 = {0, 0, 0, 0, 0, 0, 0, 0};
      __builtin_amdgcn_tensor_load_to_lds(g0, g1, gz, gz, gz8, 0);
    }

    // ---- manual stage: V tile transposed -> Vt[d][key] ----
#pragma unroll
    for (int it = 0; it < 4; ++it) {
      int q = t + it * 256;
      int row = q >> 5, c8 = (q & 31) * 8;    // row = key, c8 = d base
      bf16x8 v = *(const bf16x8*)(vbase + (size_t)(j0 + row) * QKV_LD + c8);
#pragma unroll
      for (int e = 0; e < 8; ++e) Vt[(c8 + e) * VLD + row] = v[e];
    }
    if (w == 0) __builtin_amdgcn_s_wait_tensorcnt(0);
    __syncthreads();

    // ---- S = Q @ K^T : 2 column tiles of 16 keys, 8 K-steps each ----
    f32x8 sc[2] = {};
#pragma unroll
    for (int ct = 0; ct < 2; ++ct) {
#pragma unroll
      for (int ks = 0; ks < 8; ++ks) {
        const bf16* p = &Ks[(ct * 16 + lm) * KLD + ks * 32 + half * 8];
        bf16x16 bf = cat8(*(const bf16x8*)p, *(const bf16x8*)(p + 16));
        sc[ct] = WMMA_BF16(qf[ks], bf, sc[ct]);
      }
    }
    // ---- scale + causal mask ----
#pragma unroll
    for (int ct = 0; ct < 2; ++ct) {
      int j = j0 + ct * 16 + lm;
#pragma unroll
      for (int r = 0; r < 8; ++r) {
        int i = r0w + r + half * 8;
        float v = sc[ct][r] * scale;
        sc[ct][r] = (j <= i) ? v : -3.0e38f;
      }
    }
    // ---- online softmax (row = 16-lane half reduction) ----
    float alpha[8], mrow[8];
#pragma unroll
    for (int r = 0; r < 8; ++r) {
      float mx = fmaxf(sc[0][r], sc[1][r]);
#pragma unroll
      for (int d = 1; d < 16; d <<= 1) mx = fmaxf(mx, __shfl_xor(mx, d, 32));
      float mnew = fmaxf(m_run[r], mx);
      alpha[r] = __expf(m_run[r] - mnew);
      m_run[r] = mnew;
      mrow[r] = mnew;
    }
#pragma unroll
    for (int r = 0; r < 8; ++r) {
      float p0 = __expf(sc[0][r] - mrow[r]);
      float p1 = __expf(sc[1][r] - mrow[r]);
      sc[0][r] = p0; sc[1][r] = p1;
      float rs = p0 + p1;
#pragma unroll
      for (int d = 1; d < 16; d <<= 1) rs += __shfl_xor(rs, d, 32);
      l_run[r] = l_run[r] * alpha[r] + rs;
    }
#pragma unroll
    for (int tt = 0; tt < 16; ++tt)
#pragma unroll
      for (int r = 0; r < 8; ++r) accO[tt][r] *= alpha[r];

    // ---- P (C-layout) -> per-wave LDS -> A-layout fragment ----
    bf16* pw = &Ps[w * 16 * PLD];
#pragma unroll
    for (int ct = 0; ct < 2; ++ct)
#pragma unroll
      for (int r = 0; r < 8; ++r)
        pw[(r + half * 8) * PLD + ct * 16 + lm] = (bf16)sc[ct][r];
    // LDS ops are in-order within a wave; wave-private region needs no barrier.
    bf16x16 pf;
    {
      const bf16* p = &pw[lm * PLD + half * 8];
      pf = cat8(*(const bf16x8*)p, *(const bf16x8*)(p + 16));
    }
    // ---- O += P @ V : 16 d-tiles, K-step = 32 keys ----
#pragma unroll
    for (int tt = 0; tt < 16; ++tt) {
      const bf16* p = &Vt[(tt * 16 + lm) * VLD + half * 8];
      bf16x16 vf = cat8(*(const bf16x8*)p, *(const bf16x8*)(p + 16));
      accO[tt] = WMMA_BF16(pf, vf, accO[tt]);
    }
    __syncthreads();
  }

  // ---- finalize: O / l, store bf16 attn_out[b*S+s][h*256 + d] ----
  bf16* orow = attn_out + (size_t)(b * S_LEN) * 768 + h * DH;
#pragma unroll
  for (int r = 0; r < 8; ++r) {
    float inv = 1.0f / l_run[r];
    int s = r0w + r + half * 8;
    bf16* dst = orow + (size_t)s * 768;
#pragma unroll
    for (int tt = 0; tt < 16; ++tt) dst[tt * 16 + lm] = (bf16)(accO[tt][r] * inv);
  }
}

// ---------------------------------------------------------------------------
// Launch: qkv-proj -> flash attention -> out-proj
// ---------------------------------------------------------------------------
extern "C" void kernel_launch(void* const* d_in, const int* in_sizes, int n_in,
                              void* d_out, int out_size, void* d_ws, size_t ws_size,
                              hipStream_t stream) {
  const float* X     = (const float*)d_in[0];
  const float* W_qkv = (const float*)d_in[1];
  const float* b_qkv = (const float*)d_in[2];
  const float* W_out = (const float*)d_in[3];
  const float* b_out = (const float*)d_in[4];
  float* out = (float*)d_out;

  const int B = 4, S = 2048, D = 768;
  const int M = B * S;            // 8192
  const int Nqkv = 3 * D;         // 2304

  bf16* qkv  = (bf16*)d_ws;                                            // 37.7 MB (fits L2)
  bf16* attn = (bf16*)((char*)d_ws + (size_t)M * Nqkv * sizeof(bf16)); // 12.6 MB

  dim3 blk(256);
  gemm_wmma<false, true><<<dim3(Nqkv / 128, M / 128), blk, 0, stream>>>(
      X, W_qkv, b_qkv, qkv, M, Nqkv, D);
  flash_attn<<<dim3(S / 128, B * 3), blk, 0, stream>>>(qkv, attn);
  gemm_wmma<true, false><<<dim3(D / 128, M / 128), blk, 0, stream>>>(
      attn, W_out, b_out, out, M, D, D);
}
